// DGCNN_SortPool_58076547776811
// MI455X (gfx1250) — compile-verified, matrix operand order checked by reading, not run on hardware
//
#include <hip/hip_runtime.h>
#include <hip/hip_bf16.h>
#include <math.h>

typedef __attribute__((ext_vector_type(16))) _Float16 v16h;
typedef __attribute__((ext_vector_type(8)))  float    v8f;

#define NN   102400
#define BB   256
#define MPG  400      // nodes per graph
#define EE   1638400
#define KTOP 30
#define CCAT 161
#define C1O  16
#define C2O  32

// ---------------------------------------------------------------------------
// Small elementwise helpers
// ---------------------------------------------------------------------------
__global__ void k_fill(float* p, float v, unsigned n) {
    unsigned t = blockIdx.x * blockDim.x + threadIdx.x;
    if (t < n) p[t] = v;
}

__global__ void k_deg_acc(const int* __restrict__ dst, const float* __restrict__ ew,
                          float* __restrict__ deg) {
    unsigned e = blockIdx.x * blockDim.x + threadIdx.x;
    if (e < EE) atomicAdd(&deg[dst[e]], ew[e]);
}

__global__ void k_rsqrt(const float* __restrict__ deg, float* __restrict__ dinv) {
    unsigned n = blockIdx.x * blockDim.x + threadIdx.x;
    if (n < NN) dinv[n] = rsqrtf(deg[n]);
}

__global__ void k_copy_x(const float* __restrict__ x, float* __restrict__ z) {
    unsigned t = blockIdx.x * blockDim.x + threadIdx.x;   // N*64 threads
    unsigned n = t >> 6, c = t & 63;
    if (n < NN) z[(size_t)n * CCAT + c] = x[t];
}

// ---------------------------------------------------------------------------
// Pack a row-major f32 weight W[K x Nout] (stride ldw) into the WMMA B-operand
// lane layout, f16, zero-padded to 16-wide N tiles:
//   Bp[((ntile*chunks + kc)*32 + lane)*16 + j] =
//       W[kc*32 + (lane>>4)*16 + j][ntile*16 + (lane&15)]
// One thread per output half.
// ---------------------------------------------------------------------------
__global__ void k_pack_b(const float* __restrict__ W, int ldw, int Nout,
                         int chunks, unsigned total, _Float16* __restrict__ Bp) {
    unsigned t = blockIdx.x * blockDim.x + threadIdx.x;
    if (t >= total) return;
    int j     = t & 15;
    int lane  = (t >> 4) & 31;
    unsigned rem = t >> 9;
    int kc    = rem % chunks;
    int ntile = rem / chunks;
    int col   = ntile * 16 + (lane & 15);
    int k     = kc * 32 + ((lane >> 4) << 4) + j;
    float v   = (col < Nout) ? W[(size_t)k * ldw + col] : 0.0f;
    Bp[t] = (_Float16)v;
}

// ---------------------------------------------------------------------------
// WMMA GEMM: out[M x Nout] = A[M x K] * B (+bias)(relu)
// A: f32 row-major stride lda, converted to f16 on load.
// Bp: pre-packed f16 in B-lane layout (see k_pack_b). K templated so the
// chunk loop fully unrolls into KC WMMAs with clause-able b128 loads.
// One wave computes a 16x16 tile; blockIdx.y selects the 16-wide N tile.
// ---------------------------------------------------------------------------
template <int K>
__global__ void k_wmma_gemm(const float* __restrict__ A, int lda,
                            const _Float16* __restrict__ Bp, int Nout,
                            const float* __restrict__ bias, int relu,
                            float* __restrict__ out, int ldo, int Mtiles) {
    constexpr int KC = K / 32;
    const int lane  = threadIdx.x & 31;
    const int wv    = threadIdx.x >> 5;
    const int mtile = blockIdx.x * (blockDim.x >> 5) + wv;
    if (mtile >= Mtiles) return;                 // wave-uniform exit

    const int row0  = mtile << 4;
    const int arow  = row0 + (lane & 15);
    const int khalf = (lane >> 4) << 4;          // 0 or 16
    const float* arp = A + (size_t)arow * lda + khalf;
    const _Float16* brp = Bp + ((size_t)blockIdx.y * KC * 32 + lane) * 16;

    v8f acc = {};
    #pragma unroll
    for (int kc = 0; kc < KC; ++kc) {
        v16h a;
        const float* ap = arp + kc * 32;
        #pragma unroll
        for (int j = 0; j < 16; ++j) a[j] = (_Float16)ap[j];
        v16h b = *(const v16h*)(brp + (size_t)kc * 512);
        acc = __builtin_amdgcn_wmma_f32_16x16x32_f16(
            false, a, false, b, (short)0, acc, false, false);
    }

    const int bcol = (blockIdx.y << 4) + (lane & 15);
    if (bcol < Nout) {
        const float bb = bias ? bias[bcol] : 0.0f;
        #pragma unroll
        for (int i = 0; i < 8; ++i) {
            int r = row0 + i + ((lane >> 4) << 3);   // C/D layout: M=i / M=i+8
            float v = acc[i] + bb;
            if (relu) v = fmaxf(v, 0.0f);
            out[(size_t)r * ldo + bcol] = v;
        }
    }
}

// ---------------------------------------------------------------------------
// GCN edge aggregation: 32 lanes per edge (one channel each)
// agg[dst][c] += dinv[src]*ew*dinv[dst] * h[src][c]
// ---------------------------------------------------------------------------
__global__ void k_agg32(const int* __restrict__ src, const int* __restrict__ dst,
                        const float* __restrict__ ew, const float* __restrict__ dinv,
                        const float* __restrict__ h, float* __restrict__ agg) {
    unsigned t = blockIdx.x * blockDim.x + threadIdx.x;   // EE*32 threads
    unsigned e = t >> 5, c = t & 31;
    if (e < EE) {
        int s = src[e], d = dst[e];
        float nm = dinv[s] * ew[e] * dinv[d];
        atomicAdd(&agg[(size_t)d * 32 + c], nm * h[(size_t)s * 32 + c]);
    }
}

__global__ void k_fin32(const float* __restrict__ agg, const float* __restrict__ hg,
                        const float* __restrict__ dinv, const float* __restrict__ bias,
                        float* __restrict__ z, int zoff) {
    unsigned t = blockIdx.x * blockDim.x + threadIdx.x;   // N*32 threads
    unsigned n = t >> 5, c = t & 31;
    if (n < NN) {
        float di = dinv[n];
        float v  = agg[t] + hg[t] * di * di + bias[c];
        z[(size_t)n * CCAT + zoff + c] = tanhf(v);
    }
}

// Layer 3: HID->1
__global__ void k_gemv32(const float* __restrict__ zin, const float* __restrict__ w,
                         float* __restrict__ out) {
    unsigned n = blockIdx.x * blockDim.x + threadIdx.x;
    if (n < NN) {
        const float* r = zin + (size_t)n * CCAT;
        float s = 0.0f;
        #pragma unroll
        for (int c = 0; c < 32; ++c) s += r[c] * w[c];
        out[n] = s;
    }
}

__global__ void k_agg1(const int* __restrict__ src, const int* __restrict__ dst,
                       const float* __restrict__ ew, const float* __restrict__ dinv,
                       const float* __restrict__ h, float* __restrict__ agg) {
    unsigned e = blockIdx.x * blockDim.x + threadIdx.x;
    if (e < EE) {
        int s = src[e], d = dst[e];
        atomicAdd(&agg[d], dinv[s] * ew[e] * dinv[d] * h[s]);
    }
}

__global__ void k_fin1(const float* __restrict__ agg, const float* __restrict__ hg,
                       const float* __restrict__ dinv, const float* __restrict__ bias,
                       float* __restrict__ z) {
    unsigned n = blockIdx.x * blockDim.x + threadIdx.x;
    if (n < NN) {
        float di = dinv[n];
        z[(size_t)n * CCAT + (CCAT - 1)] = agg[n] + hg[n] * di * di + bias[0];
    }
}

// ---------------------------------------------------------------------------
// SortPool: per-graph stable top-K selection on the last channel.
// rank(i) = #{j : key[j] > key[i]} + #{j < i : key[j] == key[i]}
// ---------------------------------------------------------------------------
__global__ void k_sortpool(const float* __restrict__ z, float* __restrict__ pooled) {
    __shared__ float keys[MPG];
    __shared__ int   sel[KTOP];
    const int b = blockIdx.x;
    const float* zb = z + (size_t)b * MPG * CCAT;

    for (int i = threadIdx.x; i < MPG; i += blockDim.x)
        keys[i] = zb[(size_t)i * CCAT + (CCAT - 1)];
    __syncthreads();

    for (int i = threadIdx.x; i < MPG; i += blockDim.x) {
        float k = keys[i];
        int r = 0;
        for (int j = 0; j < MPG; ++j) {
            float kj = keys[j];
            r += (kj > k) || (kj == k && j < i);
        }
        if (r < KTOP) sel[r] = i;
    }
    __syncthreads();

    for (int t = threadIdx.x; t < KTOP * CCAT; t += blockDim.x) {
        int r = t / CCAT, c = t - r * CCAT;
        pooled[((size_t)b * KTOP + r) * CCAT + c] = zb[(size_t)sel[r] * CCAT + c];
    }
}

// ---------------------------------------------------------------------------
// Conv1d stages (tiny: scalar VALU is fine)
// ---------------------------------------------------------------------------
__global__ void k_conv1(const float* __restrict__ pooled, const float* __restrict__ w,
                        const float* __restrict__ bias, float* __restrict__ y) {
    unsigned t = blockIdx.x * blockDim.x + threadIdx.x;   // B*16*26
    if (t >= BB * C1O * 26) return;
    int p = t % 26, o = (t / 26) % C1O, b = t / (26 * C1O);
    float s = bias[o];
    for (int ic = 0; ic < CCAT; ++ic) {
        const float* wp = w + ((size_t)o * CCAT + ic) * 5;
        const float* xp = pooled + ((size_t)b * KTOP + p) * CCAT + ic;
        #pragma unroll
        for (int k = 0; k < 5; ++k) s += xp[(size_t)k * CCAT] * wp[k];
    }
    y[t] = fmaxf(s, 0.0f);
}

__global__ void k_maxpool(const float* __restrict__ y, float* __restrict__ m) {
    unsigned t = blockIdx.x * blockDim.x + threadIdx.x;   // B*16*13
    if (t >= BB * C1O * 13) return;
    int q = t % 13, oc = t / 13;                           // oc = b*16+o
    const float* yp = y + (size_t)oc * 26 + 2 * q;
    m[t] = fmaxf(yp[0], yp[1]);
}

__global__ void k_conv2(const float* __restrict__ m, const float* __restrict__ w,
                        const float* __restrict__ bias, float* __restrict__ f) {
    unsigned t = blockIdx.x * blockDim.x + threadIdx.x;   // B*32*9
    if (t >= BB * C2O * 9) return;
    int p = t % 9, o = (t / 9) % C2O, b = t / (9 * C2O);
    float s = bias[o];
    for (int ic = 0; ic < C1O; ++ic) {
        const float* wp = w + ((size_t)o * C1O + ic) * 5;
        const float* xp = m + ((size_t)(b * C1O + ic)) * 13 + p;
        #pragma unroll
        for (int k = 0; k < 5; ++k) s += xp[k] * wp[k];
    }
    // reshape(B,-1) layout: f[b][o*9+p]
    f[(size_t)b * (C2O * 9) + o * 9 + p] = fmaxf(s, 0.0f);
}

// ---------------------------------------------------------------------------
extern "C" void kernel_launch(void* const* d_in, const int* in_sizes, int n_in,
                              void* d_out, int out_size, void* d_ws, size_t ws_size,
                              hipStream_t stream) {
    (void)in_sizes; (void)n_in; (void)out_size; (void)ws_size;

    const float* x      = (const float*)d_in[0];
    const int*   ei     = (const int*)d_in[1];
    const int*   src    = ei;
    const int*   dst    = ei + EE;
    const float* ew     = (const float*)d_in[3];
    const float* w0     = (const float*)d_in[4];
    const float* b0     = (const float*)d_in[5];
    const float* w1     = (const float*)d_in[6];
    const float* b1     = (const float*)d_in[7];
    const float* w2     = (const float*)d_in[8];
    const float* b2     = (const float*)d_in[9];
    const float* w3     = (const float*)d_in[10];
    const float* b3     = (const float*)d_in[11];
    const float* c1w    = (const float*)d_in[12];
    const float* c1b    = (const float*)d_in[13];
    const float* c2w    = (const float*)d_in[14];
    const float* c2b    = (const float*)d_in[15];
    const float* d1w    = (const float*)d_in[16];
    const float* d1b    = (const float*)d_in[17];
    const float* d2w    = (const float*)d_in[18];
    const float* d2b    = (const float*)d_in[19];
    float* out = (float*)d_out;

    // workspace carve-up (floats; all region sizes keep 32B alignment)
    float* ws     = (float*)d_ws;
    float* z      = ws;                                   // N*161
    float* hg     = z      + (size_t)NN * CCAT;           // N*32
    float* agg    = hg     + (size_t)NN * 32;             // N*32
    float* deg    = agg    + (size_t)NN * 32;             // N
    float* dinv   = deg    + NN;                          // N
    float* hg1    = dinv   + NN;                          // N
    float* agg1   = hg1    + NN;                          // N
    float* pooled = agg1   + NN;                          // B*30*161
    float* y1     = pooled + (size_t)BB * KTOP * CCAT;    // B*16*26
    float* m1     = y1     + (size_t)BB * C1O * 26;       // B*16*13
    float* feat   = m1     + (size_t)BB * C1O * 13;       // B*288
    float* hid    = feat   + (size_t)BB * 288;            // B*128
    // packed f16 weights (each region is a multiple of 512 halves = 1KB)
    _Float16* pb0  = (_Float16*)(hid + (size_t)BB * 128); // 2nt*2kc*512 = 2048
    _Float16* pb1  = pb0  + 2048;                         // 2nt*1kc*512 = 1024
    _Float16* pb2  = pb1  + 1024;                         // 1024
    _Float16* pbd1 = pb2  + 1024;                         // 8nt*9kc*512 = 36864
    _Float16* pbd2 = pbd1 + 36864;                        // 1nt*4kc*512 = 2048

    const dim3 blk(256);

    // ---- pack weights to f16 B-lane layout (tiny) ----
    k_pack_b<<<dim3(8),   blk, 0, stream>>>(w0,  32,  32,  2, 2048,  pb0);
    k_pack_b<<<dim3(4),   blk, 0, stream>>>(w1,  32,  32,  1, 1024,  pb1);
    k_pack_b<<<dim3(4),   blk, 0, stream>>>(w2,  32,  32,  1, 1024,  pb2);
    k_pack_b<<<dim3(144), blk, 0, stream>>>(d1w, 128, 128, 9, 36864, pbd1);
    k_pack_b<<<dim3(8),   blk, 0, stream>>>(d2w, 10,  10,  4, 2048,  pbd2);

    // ---- degrees + symmetric norm ----
    k_fill<<<dim3((NN + 255) / 256), blk, 0, stream>>>(deg, 1.0f, NN); // self-loop
    k_deg_acc<<<dim3(EE / 256), blk, 0, stream>>>(dst, ew, deg);
    k_rsqrt<<<dim3((NN + 255) / 256), blk, 0, stream>>>(deg, dinv);

    // z[:, 0:64] = x
    k_copy_x<<<dim3(NN * 64 / 256), blk, 0, stream>>>(x, z);

    const int mtilesN = NN / 16;                      // 6400
    const dim3 gN(mtilesN / 8, 2);                    // 8 waves/block, 2 N-tiles

    // ---- GCN layer 0: x[N,64] @ w0[64,32] ----
    k_wmma_gemm<64><<<gN, blk, 0, stream>>>(x, 64, pb0, 32, nullptr, 0, hg, 32, mtilesN);
    hipMemsetAsync(agg, 0, (size_t)NN * 32 * sizeof(float), stream);
    k_agg32<<<dim3(EE * 32 / 256), blk, 0, stream>>>(src, dst, ew, dinv, hg, agg);
    k_fin32<<<dim3(NN * 32 / 256), blk, 0, stream>>>(agg, hg, dinv, b0, z, 64);

    // ---- GCN layer 1: z[:,64:96] @ w1 ----
    k_wmma_gemm<32><<<gN, blk, 0, stream>>>(z + 64, CCAT, pb1, 32, nullptr, 0, hg, 32, mtilesN);
    hipMemsetAsync(agg, 0, (size_t)NN * 32 * sizeof(float), stream);
    k_agg32<<<dim3(EE * 32 / 256), blk, 0, stream>>>(src, dst, ew, dinv, hg, agg);
    k_fin32<<<dim3(NN * 32 / 256), blk, 0, stream>>>(agg, hg, dinv, b1, z, 96);

    // ---- GCN layer 2: z[:,96:128] @ w2 ----
    k_wmma_gemm<32><<<gN, blk, 0, stream>>>(z + 96, CCAT, pb2, 32, nullptr, 0, hg, 32, mtilesN);
    hipMemsetAsync(agg, 0, (size_t)NN * 32 * sizeof(float), stream);
    k_agg32<<<dim3(EE * 32 / 256), blk, 0, stream>>>(src, dst, ew, dinv, hg, agg);
    k_fin32<<<dim3(NN * 32 / 256), blk, 0, stream>>>(agg, hg, dinv, b2, z, 128);

    // ---- GCN layer 3: z[:,128:160] @ w3[32,1] (no tanh) ----
    k_gemv32<<<dim3((NN + 255) / 256), blk, 0, stream>>>(z + 128, w3, hg1);
    hipMemsetAsync(agg1, 0, (size_t)NN * sizeof(float), stream);
    k_agg1<<<dim3(EE / 256), blk, 0, stream>>>(src, dst, ew, dinv, hg1, agg1);
    k_fin1<<<dim3((NN + 255) / 256), blk, 0, stream>>>(agg1, hg1, dinv, b3, z);

    // ---- SortPool (top-30 per graph, stable) ----
    k_sortpool<<<dim3(BB), blk, 0, stream>>>(z, pooled);

    // ---- Conv stack ----
    k_conv1<<<dim3((BB * C1O * 26 + 255) / 256), blk, 0, stream>>>(pooled, c1w, c1b, y1);
    k_maxpool<<<dim3((BB * C1O * 13 + 255) / 256), blk, 0, stream>>>(y1, m1);
    k_conv2<<<dim3((BB * C2O * 9 + 255) / 256), blk, 0, stream>>>(m1, c2w, c2b, feat);

    // ---- Dense: feat[256,288]@d1w relu -> hid; hid@d2w -> out ----
    k_wmma_gemm<288><<<dim3(2, 8), blk, 0, stream>>>(feat, 288, pbd1, 128, d1b, 1, hid, 128, 16);
    k_wmma_gemm<128><<<dim3(2, 1), blk, 0, stream>>>(hid, 128, pbd2, 10, d2b, 0, out, 10, 16);
}
